// SA_2336462209657
// MI455X (gfx1250) — matrix operand order repaired and checked by reading, not compiled, not run on hardware
//
#include <hip/hip_runtime.h>
#include <hip/hip_bf16.h>

// ---------------------------------------------------------------------------
// Slot Attention forward, MI455X (gfx1250, wave32, WMMA).
// Heavy GEMMs run on v_wmma_f32_16x16x32_bf16 with double-buffered LDS tiles
// staged via GLOBAL_LOAD_ASYNC_TO_LDS_B128 (ASYNCcnt): tile i+1 streams into
// LDS while tile i feeds the matrix pipes. All B operands are pre-stored
// K-contiguous (Bp[N][K]) so A and B tiles are plain contiguous 16B chunks.
// Block tile 128x128 (8 waves, 32x64 per wave = 2x4 WMMA tiles) for high
// matrix-math per LDS byte.
// ---------------------------------------------------------------------------

typedef __attribute__((ext_vector_type(16))) __bf16 v16bf;
typedef __attribute__((ext_vector_type(8)))  __bf16 v8bf;
typedef __attribute__((ext_vector_type(8)))  float  v8f;

#define DEV static __device__ __forceinline__

DEV float sigmoidf_(float x) { return 1.0f / (1.0f + __expf(-x)); }

// Async copy 16 bytes global -> LDS (per lane).  VDST = LDS byte offset,
// VADDR = 64-bit global address.  Tracked by ASYNCcnt.
DEV void async_copy16(unsigned lds_off, const void* gptr) {
  asm volatile("global_load_async_to_lds_b128 %0, %1, off"
               :: "v"(lds_off), "v"((unsigned long long)(uintptr_t)gptr)
               : "memory");
}
DEV void wait_async0() { asm volatile("s_wait_asynccnt 0x0" ::: "memory"); }
DEV unsigned lds_off_of(const void* p) { return (unsigned)(uintptr_t)p; }

// ---------------------------------------------------------------------------
// bf16 GEMM:  C[M,N] = A[M,K] * Bp^T (+bias) (optional ReLU)
// A row-major [M,K];  Bp is the N x K (K-contiguous) operand: C = A * Bp^T.
// Requires M%128==0, N%128==0, K%32==0 (true for every call here).
// Block: 256 threads = 8 waves as 4(M) x 2(N); each wave owns 32x64 output.
// ---------------------------------------------------------------------------
__global__ __launch_bounds__(256) void gemm_bf16(
    const __bf16* __restrict__ A, const __bf16* __restrict__ Bp,
    const float* __restrict__ bias,
    float* __restrict__ Cf, __bf16* __restrict__ Cb,
    int M, int N, int K, int relu)
{
  __shared__ __align__(32) __bf16 lA[2][128 * 32];  // [row][k]  2 x 8 KB
  __shared__ __align__(32) __bf16 lB[2][128 * 32];  // [n][k]    2 x 8 KB

  const int tid  = threadIdx.x;
  const int wave = tid >> 5, lane = tid & 31;
  const int wm   = wave >> 1, wn = wave & 1;        // wave grid 4 x 2
  const int l16  = lane & 15, half = lane >> 4;
  const int bm   = blockIdx.y * 128, bn = blockIdx.x * 128;

  // this thread's staging chunks (all strides multiples of 16B)
  const int r0 = tid >> 2,          c0 = (tid & 3) << 3;   // chunk 1
  const int r1 = (tid + 256) >> 2,  c1 = c0;               // chunk 2

  v8f acc[2][4] = {};

  const int nk = K >> 5;
  // prologue: stage tile 0 into buffer 0
  async_copy16(lds_off_of(&lA[0][r0 * 32 + c0]), &A[(size_t)(bm + r0) * K + c0]);
  async_copy16(lds_off_of(&lA[0][r1 * 32 + c1]), &A[(size_t)(bm + r1) * K + c1]);
  async_copy16(lds_off_of(&lB[0][r0 * 32 + c0]), &Bp[(size_t)(bn + r0) * K + c0]);
  async_copy16(lds_off_of(&lB[0][r1 * 32 + c1]), &Bp[(size_t)(bn + r1) * K + c1]);

  for (int i = 0; i < nk; ++i) {
    wait_async0();
    __syncthreads();                       // tile i visible; tile i-1 reads done
    if (i + 1 < nk) {                      // uniform branch: stage tile i+1
      const int k1 = (i + 1) << 5, nb = (i + 1) & 1;
      async_copy16(lds_off_of(&lA[nb][r0 * 32 + c0]), &A[(size_t)(bm + r0) * K + k1 + c0]);
      async_copy16(lds_off_of(&lA[nb][r1 * 32 + c1]), &A[(size_t)(bm + r1) * K + k1 + c1]);
      async_copy16(lds_off_of(&lB[nb][r0 * 32 + c0]), &Bp[(size_t)(bn + r0) * K + k1 + c0]);
      async_copy16(lds_off_of(&lB[nb][r1 * 32 + c1]), &Bp[(size_t)(bn + r1) * K + k1 + c1]);
    }
    const __bf16* a_s = lA[i & 1];
    const __bf16* b_s = lB[i & 1];

    // fragments (ISA 16-bit A 16x32 / B 32x16 layouts)
    v16bf afrag[2], bfrag[4];
    #pragma unroll
    for (int tm = 0; tm < 2; ++tm) {
      const int r = wm * 32 + tm * 16 + l16;
      v8bf lo = *(const v8bf*)&a_s[r * 32 + half * 8];        // k = half*8 + i
      v8bf hi = *(const v8bf*)&a_s[r * 32 + 16 + half * 8];   // k = 16+half*8+i
      #pragma unroll
      for (int q = 0; q < 8; ++q) { afrag[tm][q] = lo[q]; afrag[tm][8 + q] = hi[q]; }
    }
    #pragma unroll
    for (int tn = 0; tn < 4; ++tn) {
      const int n = wn * 64 + tn * 16 + l16;
      v8bf lo = *(const v8bf*)&b_s[n * 32 + half * 8];        // k = half*8 + i
      v8bf hi = *(const v8bf*)&b_s[n * 32 + 16 + half * 8];   // k = 16+half*8+i
      #pragma unroll
      for (int q = 0; q < 8; ++q) { bfrag[tn][q] = lo[q]; bfrag[tn][8 + q] = hi[q]; }
    }

    #pragma unroll
    for (int tm = 0; tm < 2; ++tm)
      #pragma unroll
      for (int tn = 0; tn < 4; ++tn)
        acc[tm][tn] = __builtin_amdgcn_wmma_f32_16x16x32_bf16(
            false, afrag[tm], false, bfrag[tn], (short)0, acc[tm][tn],
            false, false);
  }

  // --- epilogue: C/D layout, lanes 0-15 rows r, lanes 16-31 rows r+8 ---
  #pragma unroll
  for (int tm = 0; tm < 2; ++tm) {
    #pragma unroll
    for (int tn = 0; tn < 4; ++tn) {
      const int col = bn + wn * 64 + tn * 16 + l16;
      const float bv = bias ? bias[col] : 0.0f;
      #pragma unroll
      for (int r = 0; r < 8; ++r) {
        const int row = bm + wm * 32 + tm * 16 + half * 8 + r;
        float v = acc[tm][tn][r] + bv;
        if (relu) v = fmaxf(v, 0.0f);
        if (Cf) Cf[(size_t)row * N + col] = v;
        if (Cb) Cb[(size_t)row * N + col] = (__bf16)v;
      }
    }
  }
}

// ---------------------------------------------------------------------------
// LayerNorm over last dim (wave per row), f32 in -> bf16 out.
// ---------------------------------------------------------------------------
__global__ __launch_bounds__(256) void ln_rows(
    const float* __restrict__ x, const float* __restrict__ g,
    const float* __restrict__ b, __bf16* __restrict__ y, int C, int rows)
{
  const int wave = threadIdx.x >> 5, lane = threadIdx.x & 31;
  const int row  = blockIdx.x * 8 + wave;
  if (row >= rows) return;
  const float* xr = x + (size_t)row * C;
  float s = 0.0f, s2 = 0.0f;
  for (int c = lane; c < C; c += 32) { float v = xr[c]; s += v; s2 += v * v; }
  #pragma unroll
  for (int off = 16; off > 0; off >>= 1) {
    s  += __shfl_xor(s,  off, 32);
    s2 += __shfl_xor(s2, off, 32);
  }
  const float mean = s / C;
  const float inv  = rsqrtf(s2 / C - mean * mean + 1e-5f);
  __bf16* yr = y + (size_t)row * C;
  for (int c = lane; c < C; c += 32)
    yr[c] = (__bf16)((xr[c] - mean) * inv * g[c] + b[c]);
}

// ---------------------------------------------------------------------------
// small helpers
// ---------------------------------------------------------------------------
__global__ void cvt_bf16(const float* __restrict__ s, __bf16* __restrict__ d, int n) {
  int i = blockIdx.x * 256 + threadIdx.x;
  if (i < n) d[i] = (__bf16)s[i];
}
// s[R,C] -> d[C,R]
__global__ void cvt_bf16_T(const float* __restrict__ s, __bf16* __restrict__ d,
                           int R, int C) {
  int i = blockIdx.x * 256 + threadIdx.x;
  if (i < R * C) { int r = i / C, c = i % C; d[(size_t)c * R + r] = (__bf16)s[i]; }
}
__global__ void bcast_slots(const float* __restrict__ si, float* __restrict__ slots, int n) {
  int i = blockIdx.x * 256 + threadIdx.x;
  if (i < n) slots[i] = si[i & 2047];      // S*D = 2048
}
__global__ void addv(float* __restrict__ a, const float* __restrict__ b, int n) {
  int i = blockIdx.x * 256 + threadIdx.x;
  if (i < n) a[i] += b[i];
}
__global__ void copyv(const float* __restrict__ a, float* __restrict__ o, int n) {
  int i = blockIdx.x * 256 + threadIdx.x;
  if (i < n) o[i] = a[i];
}

// ---------------------------------------------------------------------------
// dots + softmax over the SLOT dim (S=8). One thread per token.
// attn written pre-normalized-over-t (softmax_s + EPS).
// ---------------------------------------------------------------------------
__global__ __launch_bounds__(256) void attn_dots(
    const __bf16* __restrict__ q,      // [B, 8, 256]
    const __bf16* __restrict__ keys,   // [B, N, 256]
    float* __restrict__ attn,          // [B, 8, N]
    int Ntok)
{
  __shared__ __align__(16) __bf16 lq[8 * 256];
  const int tiles = Ntok >> 8;
  const int b  = blockIdx.x / tiles;
  const int t0 = (blockIdx.x % tiles) << 8;
  *(v8bf*)&lq[threadIdx.x * 8] =
      *(const v8bf*)&q[(size_t)b * 2048 + threadIdx.x * 8];
  __syncthreads();

  const int t = t0 + threadIdx.x;
  const __bf16* kr = keys + ((size_t)b * Ntok + t) * 256;
  float dot[8] = {};
  for (int c = 0; c < 256; c += 8) {
    v8bf kv = *(const v8bf*)&kr[c];
    float kf[8];
    #pragma unroll
    for (int j = 0; j < 8; ++j) kf[j] = (float)kv[j];
    #pragma unroll
    for (int s = 0; s < 8; ++s) {
      v8bf qv = *(const v8bf*)&lq[s * 256 + c];
      #pragma unroll
      for (int j = 0; j < 8; ++j) dot[s] += (float)qv[j] * kf[j];
    }
  }
  const float SCALE = 0.0625f;   // 256^-0.5
  float mx = -1e30f;
  #pragma unroll
  for (int s = 0; s < 8; ++s) { dot[s] *= SCALE; mx = fmaxf(mx, dot[s]); }
  float ex[8], sum = 0.0f;
  #pragma unroll
  for (int s = 0; s < 8; ++s) { ex[s] = __expf(dot[s] - mx); sum += ex[s]; }
  const float isum = 1.0f / sum;
  #pragma unroll
  for (int s = 0; s < 8; ++s)
    attn[((size_t)b * 8 + s) * Ntok + t] = ex[s] * isum + 1e-8f;
}

// sum over t per (b,s) row — deterministic block reduction
__global__ __launch_bounds__(256) void attn_rowsum(
    const float* __restrict__ attn, float* __restrict__ rowsum, int Ntok)
{
  __shared__ float red[256];
  const int row = blockIdx.x;
  float s = 0.0f;
  for (int t = threadIdx.x; t < Ntok; t += 256) s += attn[(size_t)row * Ntok + t];
  red[threadIdx.x] = s; __syncthreads();
  for (int off = 128; off > 0; off >>= 1) {
    if (threadIdx.x < off) red[threadIdx.x] += red[threadIdx.x + off];
    __syncthreads();
  }
  if (threadIdx.x == 0) rowsum[row] = red[0];
}

// updates[b,s,:] = sum_t (attn[b,s,t]/rowsum[b,s]) * values[b,t,:]
__global__ __launch_bounds__(256) void attn_updates(
    const float* __restrict__ attn, const float* __restrict__ rowsum,
    const __bf16* __restrict__ values, __bf16* __restrict__ upd, int Ntok)
{
  __shared__ float la[8 * 128];
  __shared__ float inv[8];
  const int b = blockIdx.x, d = threadIdx.x;
  if (threadIdx.x < 8) inv[threadIdx.x] = 1.0f / rowsum[b * 8 + threadIdx.x];
  __syncthreads();
  float acc[8] = {};
  for (int t0 = 0; t0 < Ntok; t0 += 128) {
    __syncthreads();
    for (int c = threadIdx.x; c < 1024; c += 256) {
      const int s = c >> 7, i = c & 127;
      la[c] = attn[((size_t)b * 8 + s) * Ntok + t0 + i] * inv[s];
    }
    __syncthreads();
    for (int i = 0; i < 128; ++i) {
      const float v = (float)values[((size_t)b * Ntok + t0 + i) * 256 + d];
      #pragma unroll
      for (int s = 0; s < 8; ++s) acc[s] += la[s * 128 + i] * v;
    }
  }
  #pragma unroll
  for (int s = 0; s < 8; ++s)
    upd[((size_t)b * 8 + s) * 256 + d] = (__bf16)acc[s];
}

// GRU gate combine, torch semantics (r,z,n order); slots updated in place.
__global__ void gru_combine(const float* __restrict__ gi,
                            const float* __restrict__ gh,
                            float* __restrict__ slots, int n)
{
  int i = blockIdx.x * 256 + threadIdx.x;
  if (i >= n) return;
  const int row = i >> 8, d = i & 255;
  const size_t base = (size_t)row * 768;
  const float r = sigmoidf_(gi[base + d]       + gh[base + d]);
  const float z = sigmoidf_(gi[base + 256 + d] + gh[base + 256 + d]);
  const float nn = tanhf(gi[base + 512 + d] + r * gh[base + 512 + d]);
  const float hp = slots[i];
  slots[i] = (1.0f - z) * nn + z * hp;
}

// ---------------------------------------------------------------------------
extern "C" void kernel_launch(void* const* d_in, const int* in_sizes, int n_in,
                              void* d_out, int out_size, void* d_ws, size_t ws_size,
                              hipStream_t stream) {
  constexpr int Bsz = 32, Ntok = 4096, DIN = 768, D = 256, S = 8, ITERS = 3;
  constexpr int Mtok = Bsz * Ntok;            // 131072
  constexpr int BS = Bsz * S;                 // 256
  constexpr int SL = BS * D;                  // 65536 slot elements

  const float* inputs = (const float*)d_in[0];
  const float* sinit  = (const float*)d_in[1];
  const float* ln1g = (const float*)d_in[2];  const float* ln1b = (const float*)d_in[3];
  const float* w1   = (const float*)d_in[4];  const float* b1   = (const float*)d_in[5];
  const float* w2   = (const float*)d_in[6];  const float* b2   = (const float*)d_in[7];
  const float* ln2g = (const float*)d_in[8];  const float* ln2b = (const float*)d_in[9];
  const float* wq   = (const float*)d_in[10]; const float* wk   = (const float*)d_in[11];
  const float* wv   = (const float*)d_in[12];
  const float* normg= (const float*)d_in[13]; const float* normb= (const float*)d_in[14];
  const float* wih  = (const float*)d_in[15]; const float* whh  = (const float*)d_in[16];
  const float* bih  = (const float*)d_in[17]; const float* bhh  = (const float*)d_in[18];
  const float* mlng = (const float*)d_in[19]; const float* mlnb = (const float*)d_in[20];
  const float* mw1  = (const float*)d_in[21]; const float* mb1  = (const float*)d_in[22];
  const float* mw2  = (const float*)d_in[23]; const float* mb2  = (const float*)d_in[24];

  char* ws = (char*)d_ws;
  size_t off = 0;
  auto alloc = [&](size_t bytes) { size_t o = off; off = (off + bytes + 255) & ~(size_t)255; return ws + o; };

  // Two big regions, reused across phases:
  //   bufA: xln bf16 (201MB) -> h2 f32 (134MB) -> keys+values bf16 (134MB)
  //   bufB: h1 bf16 (201MB)  -> feats bf16 (67MB)
  char* bufA = (char*)alloc((size_t)Mtok * DIN * 2);
  char* bufB = (char*)alloc((size_t)Mtok * DIN * 2);
  __bf16* xln   = (__bf16*)bufA;
  __bf16* h1    = (__bf16*)bufB;
  float*  h2    = (float*)bufA;
  __bf16* feats = (__bf16*)bufB;
  __bf16* keysb = (__bf16*)bufA;
  __bf16* valsb = (__bf16*)(bufA + (size_t)Mtok * D * 2);

  float*  slots   = (float*) alloc(SL * 4);
  __bf16* slotsB  = (__bf16*)alloc(SL * 2);
  __bf16* slotsLN = (__bf16*)alloc(SL * 2);
  __bf16* qb      = (__bf16*)alloc(SL * 2);
  float*  attn    = (float*) alloc((size_t)Bsz * S * Ntok * 4);
  float*  rowsum  = (float*) alloc(BS * 4);
  __bf16* updb    = (__bf16*)alloc(SL * 2);
  float*  gi      = (float*) alloc((size_t)BS * 3 * D * 4);
  float*  gh      = (float*) alloc((size_t)BS * 3 * D * 4);
  __bf16* mln     = (__bf16*)alloc(SL * 2);
  __bf16* m1b     = (__bf16*)alloc((size_t)BS * 4 * D * 2);
  float*  m2      = (float*) alloc(SL * 4);
  // B operands, ALL stored K-contiguous as Bp[N][K]
  __bf16* w1p     = (__bf16*)alloc((size_t)DIN * DIN * 2);   // [768][768]
  __bf16* w2p     = (__bf16*)alloc((size_t)D * DIN * 2);     // [256][768]
  __bf16* wqp     = (__bf16*)alloc((size_t)D * D * 2);       // [256][256]
  __bf16* wkp     = (__bf16*)alloc((size_t)D * D * 2);
  __bf16* wvp     = (__bf16*)alloc((size_t)D * D * 2);
  __bf16* wihp    = (__bf16*)alloc((size_t)3 * D * D * 2);   // [768][256]
  __bf16* whhp    = (__bf16*)alloc((size_t)3 * D * D * 2);
  __bf16* mw1p    = (__bf16*)alloc((size_t)4 * D * D * 2);   // [1024][256]
  __bf16* mw2p    = (__bf16*)alloc((size_t)D * 4 * D * 2);   // [256][1024]

  // --- weight conversion to bf16, K-contiguous Bp layout ---
  // ref uses X@W       -> Bp = W^T  (transposing convert)
  // ref uses X@W.T     -> Bp = W    (plain convert)
  cvt_bf16_T<<<DIN * DIN / 256, 256, 0, stream>>>(w1, w1p, DIN, DIN);
  cvt_bf16_T<<<DIN * D / 256, 256, 0, stream>>>(w2, w2p, DIN, D);
  cvt_bf16<<<D * D / 256, 256, 0, stream>>>(wq, wqp, D * D);
  cvt_bf16<<<D * D / 256, 256, 0, stream>>>(wk, wkp, D * D);
  cvt_bf16<<<D * D / 256, 256, 0, stream>>>(wv, wvp, D * D);
  cvt_bf16<<<3 * D * D / 256, 256, 0, stream>>>(wih, wihp, 3 * D * D);
  cvt_bf16<<<3 * D * D / 256, 256, 0, stream>>>(whh, whhp, 3 * D * D);
  cvt_bf16_T<<<D * 4 * D / 256, 256, 0, stream>>>(mw1, mw1p, D, 4 * D);
  cvt_bf16_T<<<4 * D * D / 256, 256, 0, stream>>>(mw2, mw2p, 4 * D, D);

  // --- feature pipeline (the 240 GFLOP part) ---
  ln_rows<<<Mtok / 8, 256, 0, stream>>>(inputs, ln1g, ln1b, xln, DIN, Mtok);
  gemm_bf16<<<dim3(DIN / 128, Mtok / 128), 256, 0, stream>>>(
      xln, w1p, b1, nullptr, h1, Mtok, DIN, DIN, 1);
  gemm_bf16<<<dim3(D / 128, Mtok / 128), 256, 0, stream>>>(
      h1, w2p, b2, h2, nullptr, Mtok, D, DIN, 0);
  ln_rows<<<Mtok / 8, 256, 0, stream>>>(h2, ln2g, ln2b, feats, D, Mtok);
  gemm_bf16<<<dim3(D / 128, Mtok / 128), 256, 0, stream>>>(
      feats, wkp, nullptr, nullptr, keysb, Mtok, D, D, 0);
  gemm_bf16<<<dim3(D / 128, Mtok / 128), 256, 0, stream>>>(
      feats, wvp, nullptr, nullptr, valsb, Mtok, D, D, 0);

  // --- slot iteration (straight-through is forward-identity; skip it) ---
  bcast_slots<<<SL / 256, 256, 0, stream>>>(sinit, slots, SL);
  for (int it = 0; it < ITERS; ++it) {
    cvt_bf16<<<SL / 256, 256, 0, stream>>>(slots, slotsB, SL);           // slots_prev
    ln_rows<<<BS / 8, 256, 0, stream>>>(slots, normg, normb, slotsLN, D, BS);
    gemm_bf16<<<dim3(D / 128, BS / 128), 256, 0, stream>>>(
        slotsLN, wqp, nullptr, nullptr, qb, BS, D, D, 0);
    attn_dots<<<Bsz * (Ntok / 256), 256, 0, stream>>>(qb, keysb, attn, Ntok);
    attn_rowsum<<<Bsz * S, 256, 0, stream>>>(attn, rowsum, Ntok);
    attn_updates<<<Bsz, 256, 0, stream>>>(attn, rowsum, valsb, updb, Ntok);
    gemm_bf16<<<dim3(3 * D / 128, BS / 128), 256, 0, stream>>>(
        updb, wihp, bih, gi, nullptr, BS, 3 * D, D, 0);
    gemm_bf16<<<dim3(3 * D / 128, BS / 128), 256, 0, stream>>>(
        slotsB, whhp, bhh, gh, nullptr, BS, 3 * D, D, 0);
    gru_combine<<<SL / 256, 256, 0, stream>>>(gi, gh, slots, SL);
    ln_rows<<<BS / 8, 256, 0, stream>>>(slots, mlng, mlnb, mln, D, BS);
    gemm_bf16<<<dim3(4 * D / 128, BS / 128), 256, 0, stream>>>(
        mln, mw1p, mb1, nullptr, m1b, BS, 4 * D, D, 1);
    gemm_bf16<<<dim3(D / 128, BS / 128), 256, 0, stream>>>(
        m1b, mw2p, mb2, m2, nullptr, BS, D, 4 * D, 0);
    addv<<<SL / 256, 256, 0, stream>>>(slots, m2, SL);
  }

  copyv<<<SL / 256, 256, 0, stream>>>(slots, (float*)d_out, SL);
}
